// NTMCell_62277025792416
// MI455X (gfx1250) — matrix operand order repaired and checked by reading, not compiled, not run on hardware
//
#include <hip/hip_runtime.h>
#include <hip/hip_bf16.h>

// Problem constants (match reference)
#define B_     8192
#define N_     128
#define M_     64
#define C_     256
#define DIN_   128
#define DOUT_  128
#define CTRL_IN_ 192   // DIN + M
#define EPS_   1e-16f

// Padded head-projection widths (tile multiples of 16)
#define RO_PAD 80     // 70  -> 80
#define WO_PAD 208    // 198 -> 208

typedef __attribute__((ext_vector_type(16))) _Float16 v16h;
typedef __attribute__((ext_vector_type(8)))  _Float16 v8h;
typedef __attribute__((ext_vector_type(8)))  float    v8f;

// ---------------------------------------------------------------------------
// Math helpers
// ---------------------------------------------------------------------------
__device__ __forceinline__ float sigmoidf_(float x) { return 1.0f / (1.0f + __expf(-x)); }
__device__ __forceinline__ float softplusf_(float x) {
    return (x > 20.0f) ? x : log1pf(__expf(x));
}

// ---------------------------------------------------------------------------
// WMMA fragment helpers (CDNA5 wave32, V_WMMA_F32_16X16X32_F16)
//
// A (16xK f16, ISA 7.12.2): lanes 0-15 -> row = lane,    K in {0..7,16..23}
//                           lanes 16-31 -> row = lane-16, K in {8..15,24..31}
//   -> per lane: 16 contiguous halves at kbase + hi*16 (k = 16*hi + e, same order as B)
// B (Kx16 f16): col = lane%16, K = e + (lane>=16 ? 16 : 0) in this code (matches A; not ISA order)
//   -> per lane: 16 contiguous halves of weight row j at kbase + hi*16
// C/D (16x16 f32): col = lane%16, VGPR i -> row = i + (lane>=16 ? 8 : 0)
// ---------------------------------------------------------------------------
__device__ __forceinline__ v16h load_a_frag(const _Float16* A, int lda, int kbase, int lane) {
    int row = lane & 15;
    int hi  = lane >> 4;
    const _Float16* p = A + row * lda + kbase + hi * 16;
    v8h lo = *(const v8h*)(p);          // K = kbase+hi*16 .. +7
    v8h hiv = *(const v8h*)(p + 8);     // K = kbase+hi*16+8 .. +15
    v16h a;
#pragma unroll
    for (int e = 0; e < 8; ++e) { a[e] = lo[e]; a[e + 8] = hiv[e]; }
    return a;
}

// B fragment from a pre-converted f16 weight matrix (row-major, zero-padded rows).
__device__ __forceinline__ v16h load_b_frag(const _Float16* __restrict__ Wh, int ldw,
                                            int jbase, int kbase, int lane) {
    int col = lane & 15;
    int hi  = lane >> 4;
    return *(const v16h*)(Wh + (size_t)(jbase + col) * ldw + kbase + hi * 16);
}

__device__ __forceinline__ v8f wmma_acc(v16h a, v16h b, v8f c) {
    return __builtin_amdgcn_wmma_f32_16x16x32_f16(
        /*neg_a=*/false, a, /*neg_b=*/false, b,
        /*c_mod=*/(short)0, c, /*reuse_a=*/false, /*reuse_b=*/false);
}

// ---------------------------------------------------------------------------
// Kernel 0: one-shot weight conversion fp32 -> f16 (zero-padded ragged rows),
// plus zero-padded biases.  Tiny (~0.5 M elements), runs once per call.
// ---------------------------------------------------------------------------
__global__ __launch_bounds__(256) void k_cvt(
    const float* __restrict__ W_ih, const float* __restrict__ W_hh,
    const float* __restrict__ read_W, const float* __restrict__ read_b,
    const float* __restrict__ write_W, const float* __restrict__ write_b,
    const float* __restrict__ fc_W,
    _Float16* __restrict__ Wih_h, _Float16* __restrict__ Whh_h,
    _Float16* __restrict__ readW_h, _Float16* __restrict__ writeW_h,
    _Float16* __restrict__ fcW_h,
    float* __restrict__ read_b_p, float* __restrict__ write_b_p)
{
    int tid = blockIdx.x * blockDim.x + threadIdx.x;
    int nth = gridDim.x * blockDim.x;
    for (int i = tid; i < 3 * C_ * CTRL_IN_; i += nth) Wih_h[i] = (_Float16)W_ih[i];
    for (int i = tid; i < 3 * C_ * C_;       i += nth) Whh_h[i] = (_Float16)W_hh[i];
    for (int i = tid; i < RO_PAD * C_; i += nth) {
        int r = i >> 8;   // / 256
        readW_h[i] = (r < M_ + 6) ? (_Float16)read_W[i] : (_Float16)0.0f;
    }
    for (int i = tid; i < WO_PAD * C_; i += nth) {
        int r = i >> 8;
        writeW_h[i] = (r < 3 * M_ + 6) ? (_Float16)write_W[i] : (_Float16)0.0f;
    }
    for (int i = tid; i < DOUT_ * (C_ + M_); i += nth) fcW_h[i] = (_Float16)fc_W[i];
    for (int i = tid; i < RO_PAD; i += nth) read_b_p[i]  = (i < M_ + 6)     ? read_b[i]  : 0.0f;
    for (int i = tid; i < WO_PAD; i += nth) write_b_p[i] = (i < 3 * M_ + 6) ? write_b[i] : 0.0f;
}

// ---------------------------------------------------------------------------
// Kernel 1: GRU controller.  16 batch rows per block, 256 threads (8 waves).
// Each wave owns two 16x16 output tiles of h; all six gate fragments live in
// registers so the GRU nonlinearity fuses directly after the WMMAs.
// ---------------------------------------------------------------------------
__global__ __launch_bounds__(256) void k_controller(
    const float* __restrict__ x, const float* __restrict__ prev_read,
    const float* __restrict__ h_prev,
    const _Float16* __restrict__ Wih_h, const float* __restrict__ b_ih,
    const _Float16* __restrict__ Whh_h, const float* __restrict__ b_hh,
    float* __restrict__ h_out)
{
    constexpr int IN_LD = 200;   // 192 + pad (mult of 8)
    constexpr int H_LD  = 264;   // 256 + pad (mult of 8)
    __shared__ _Float16 sIn[16 * IN_LD];
    __shared__ _Float16 sH [16 * H_LD];

    int tid = threadIdx.x;
    int b0  = blockIdx.x * 16;

    for (int idx = tid; idx < 16 * CTRL_IN_; idx += 256) {
        int r = idx / CTRL_IN_, c = idx % CTRL_IN_;
        float v = (c < DIN_) ? x[(size_t)(b0 + r) * DIN_ + c]
                             : prev_read[(size_t)(b0 + r) * M_ + (c - DIN_)];
        sIn[r * IN_LD + c] = (_Float16)v;
    }
    for (int idx = tid; idx < 16 * C_; idx += 256) {
        int r = idx >> 8, c = idx & 255;
        sH[r * H_LD + c] = (_Float16)h_prev[(size_t)(b0 + r) * C_ + c];
    }
    __syncthreads();

    int wv = tid >> 5, lane = tid & 31;
    int col = lane & 15, hi = lane >> 4;

    for (int t = 0; t < 2; ++t) {
        int jt = (wv + t * 8) * 16;                 // output-column base in [0,256)
        v8f air = {}, aiz = {}, ain = {}, ahr = {}, ahz = {}, ahn = {};
        for (int kk = 0; kk < CTRL_IN_; kk += 32) {
            v16h a = load_a_frag(sIn, IN_LD, kk, lane);
            air = wmma_acc(a, load_b_frag(Wih_h, CTRL_IN_, jt,          kk, lane), air);
            aiz = wmma_acc(a, load_b_frag(Wih_h, CTRL_IN_, jt + C_,     kk, lane), aiz);
            ain = wmma_acc(a, load_b_frag(Wih_h, CTRL_IN_, jt + 2 * C_, kk, lane), ain);
        }
        for (int kk = 0; kk < C_; kk += 32) {
            v16h a = load_a_frag(sH, H_LD, kk, lane);
            ahr = wmma_acc(a, load_b_frag(Whh_h, C_, jt,          kk, lane), ahr);
            ahz = wmma_acc(a, load_b_frag(Whh_h, C_, jt + C_,     kk, lane), ahz);
            ahn = wmma_acc(a, load_b_frag(Whh_h, C_, jt + 2 * C_, kk, lane), ahn);
        }
        int j = jt + col;
        float bir = b_ih[j], biz = b_ih[C_ + j], bin = b_ih[2 * C_ + j];
        float bhr = b_hh[j], bhz = b_hh[C_ + j], bhn = b_hh[2 * C_ + j];
#pragma unroll
        for (int i = 0; i < 8; ++i) {
            int row = i + hi * 8;
            float rg = sigmoidf_((air[i] + bir) + (ahr[i] + bhr));
            float zg = sigmoidf_((aiz[i] + biz) + (ahz[i] + bhz));
            float ng = tanhf((ain[i] + bin) + rg * (ahn[i] + bhn));
            float hp = (float)sH[row * H_LD + j];
            h_out[(size_t)(b0 + row) * C_ + j] = (1.0f - zg) * ng + zg * hp;
        }
    }
}

// ---------------------------------------------------------------------------
// Kernel 2: read/write head projections into padded workspace arrays.
// 18 output tiles (5 read + 13 write) over 8 waves; weights pre-padded so
// there are no guards anywhere.
// ---------------------------------------------------------------------------
__global__ __launch_bounds__(256) void k_heads(
    const float* __restrict__ h_in,
    const _Float16* __restrict__ readW_h,  const float* __restrict__ read_b_p,
    const _Float16* __restrict__ writeW_h, const float* __restrict__ write_b_p,
    float* __restrict__ ro_s, float* __restrict__ wo_s)
{
    constexpr int H_LD = 264;
    __shared__ _Float16 sH[16 * H_LD];

    int tid = threadIdx.x;
    int b0  = blockIdx.x * 16;

    for (int idx = tid; idx < 16 * C_; idx += 256) {
        int r = idx >> 8, c = idx & 255;
        sH[r * H_LD + c] = (_Float16)h_in[(size_t)(b0 + r) * C_ + c];
    }
    __syncthreads();

    int wv = tid >> 5, lane = tid & 31;
    int col = lane & 15, hi = lane >> 4;

    for (int t = wv; t < 18; t += 8) {
        bool isRead = (t < 5);
        const _Float16* W = isRead ? readW_h  : writeW_h;
        const float* bias = isRead ? read_b_p : write_b_p;
        int   ostride     = isRead ? RO_PAD : WO_PAD;
        float* outp       = isRead ? ro_s : wo_s;
        int   jt          = (isRead ? t : t - 5) * 16;

        v8f acc = {};
        for (int kk = 0; kk < C_; kk += 32) {
            v16h a = load_a_frag(sH, H_LD, kk, lane);
            acc = wmma_acc(a, load_b_frag(W, C_, jt, kk, lane), acc);
        }
        int j = jt + col;
        float bv = bias[j];
#pragma unroll
        for (int i = 0; i < 8; ++i) {
            int row = i + hi * 8;
            outp[(size_t)(b0 + row) * ostride + j] = acc[i] + bv;
        }
    }
}

// ---------------------------------------------------------------------------
// Kernel 3: addressing + read + memory update.  One block (128 threads) per
// sample; the 128x64 memory tile is staged into LDS with CDNA5 async
// LDS-DMA (GLOBAL_LOAD_ASYNC_TO_LDS_B128, tracked by ASYNCcnt), stride 65
// padding keeps all row accesses bank-conflict-free.
// ---------------------------------------------------------------------------
__device__ __forceinline__ float red_max128(float v, float* red, int tid) {
    red[tid] = v; __syncthreads();
    for (int o = 64; o > 0; o >>= 1) {
        if (tid < o) red[tid] = fmaxf(red[tid], red[tid + o]);
        __syncthreads();
    }
    float r = red[0]; __syncthreads();
    return r;
}
__device__ __forceinline__ float red_sum128(float v, float* red, int tid) {
    red[tid] = v; __syncthreads();
    for (int o = 64; o > 0; o >>= 1) {
        if (tid < o) red[tid] = red[tid] + red[tid + o];
        __syncthreads();
    }
    float r = red[0]; __syncthreads();
    return r;
}

__global__ __launch_bounds__(128) void k_memory(
    const float* __restrict__ memory,
    const float* __restrict__ w_read_prev, const float* __restrict__ w_write_prev,
    const float* __restrict__ ro_s, const float* __restrict__ wo_s,
    float* __restrict__ r_out, float* __restrict__ wr_out,
    float* __restrict__ ww_out, float* __restrict__ mem_out)
{
    constexpr int MLD = 65;
    __shared__ float s_mem[N_ * MLD];
    __shared__ float s_rk[M_], s_wk[M_], s_e[M_], s_a[M_];
    __shared__ float s_red[N_];
    __shared__ float s_wgR[N_], s_wgW[N_];
    __shared__ float s_wr[N_],  s_ww[N_];
    // 0:rbeta 1:rg 2-4:rs 5:rgamma 6:|rk|  7:wbeta 8:wg 9-11:ws 12:wgamma 13:|wk|
    __shared__ float s_sc[14];

    int tid = threadIdx.x;
    size_t b = blockIdx.x;
    const float* memb = memory + b * (size_t)(N_ * M_);

    // Async LDS-DMA: each lane moves one 16B chunk (4 floats) per op; rows are
    // 64 floats so a chunk never crosses the stride-65 padding.  Generic LDS
    // pointers carry the LDS byte offset in addr[31:0] (flat aperture rule).
    {
        unsigned lds_base = (unsigned)(unsigned long long)(&s_mem[0]);
#pragma unroll
        for (int i = 0; i < 16; ++i) {
            int idx4 = i * 512 + tid * 4;                 // first float of this chunk
            unsigned lds_off = lds_base + (unsigned)(((idx4 >> 6) * MLD + (idx4 & 63)) * 4);
            const float* gp = memb + idx4;
            asm volatile("global_load_async_to_lds_b128 %0, %1, off"
                         :: "v"(lds_off), "v"(gp) : "memory");
        }
        asm volatile("s_wait_asynccnt 0" ::: "memory");
    }

    const float* rob = ro_s + b * RO_PAD;
    const float* wob = wo_s + b * WO_PAD;
    if (tid < M_) {
        s_rk[tid] = rob[tid];
        s_wk[tid] = wob[tid];
        s_e[tid]  = sigmoidf_(wob[M_ + 6 + tid]);       // erase
        s_a[tid]  = wob[2 * M_ + 6 + tid];              // add
    }
    __syncthreads();

    if (tid == 0) {
        s_sc[0] = softplusf_(rob[M_]);
        s_sc[1] = sigmoidf_(rob[M_ + 1]);
        float m0 = fmaxf(fmaxf(rob[M_ + 2], rob[M_ + 3]), rob[M_ + 4]);
        float e0 = __expf(rob[M_ + 2] - m0), e1 = __expf(rob[M_ + 3] - m0), e2 = __expf(rob[M_ + 4] - m0);
        float es = e0 + e1 + e2;
        s_sc[2] = e0 / es; s_sc[3] = e1 / es; s_sc[4] = e2 / es;
        s_sc[5] = 1.0f + softplusf_(rob[M_ + 5]);
        float ss = 0.0f;
        for (int m = 0; m < M_; ++m) ss += s_rk[m] * s_rk[m];
        s_sc[6] = sqrtf(ss);
    }
    if (tid == 1) {
        s_sc[7] = softplusf_(wob[M_]);
        s_sc[8] = sigmoidf_(wob[M_ + 1]);
        float m0 = fmaxf(fmaxf(wob[M_ + 2], wob[M_ + 3]), wob[M_ + 4]);
        float e0 = __expf(wob[M_ + 2] - m0), e1 = __expf(wob[M_ + 3] - m0), e2 = __expf(wob[M_ + 4] - m0);
        float es = e0 + e1 + e2;
        s_sc[9] = e0 / es; s_sc[10] = e1 / es; s_sc[11] = e2 / es;
        s_sc[12] = 1.0f + softplusf_(wob[M_ + 5]);
        float ss = 0.0f;
        for (int m = 0; m < M_; ++m) ss += s_wk[m] * s_wk[m];
        s_sc[13] = sqrtf(ss);
    }
    __syncthreads();

    // Cosine-similarity scores for memory row `tid`
    float rdot = 0.0f, wdot = 0.0f, nn = 0.0f;
    const float* mrow = &s_mem[tid * MLD];
#pragma unroll 8
    for (int m = 0; m < M_; ++m) {
        float v = mrow[m];
        rdot += s_rk[m] * v;
        wdot += s_wk[m] * v;
        nn   += v * v;
    }
    float normn = sqrtf(nn);
    float scR = s_sc[0] * rdot / (s_sc[6]  * normn + EPS_);
    float scW = s_sc[7] * wdot / (s_sc[13] * normn + EPS_);

    // softmax + interpolation (read head)
    float mx = red_max128(scR, s_red, tid);
    float ex = __expf(scR - mx);
    float sm = red_sum128(ex, s_red, tid);
    float gR = s_sc[1];
    s_wgR[tid] = gR * (ex / sm) + (1.0f - gR) * w_read_prev[b * N_ + tid];

    // softmax + interpolation (write head)
    mx = red_max128(scW, s_red, tid);
    ex = __expf(scW - mx);
    sm = red_sum128(ex, s_red, tid);
    float gW = s_sc[8];
    s_wgW[tid] = gW * (ex / sm) + (1.0f - gW) * w_write_prev[b * N_ + tid];
    __syncthreads();

    // circular shift + sharpen + renormalize
    float wsR = s_sc[2] * s_wgR[(tid + 127) & 127] + s_sc[3] * s_wgR[tid] + s_sc[4] * s_wgR[(tid + 1) & 127];
    float wpR = powf(fmaxf(wsR, 0.0f), s_sc[5]);
    float sR  = red_sum128(wpR, s_red, tid);
    float wR  = wpR / (sR + EPS_);
    s_wr[tid] = wR;
    wr_out[b * N_ + tid] = wR;

    float wsW = s_sc[9] * s_wgW[(tid + 127) & 127] + s_sc[10] * s_wgW[tid] + s_sc[11] * s_wgW[(tid + 1) & 127];
    float wpW = powf(fmaxf(wsW, 0.0f), s_sc[12]);
    float sW  = red_sum128(wpW, s_red, tid);
    float wW  = wpW / (sW + EPS_);
    s_ww[tid] = wW;
    ww_out[b * N_ + tid] = wW;
    __syncthreads();

    // read vector r[m] = sum_n w_read[n] * mem[n][m]
    if (tid < M_) {
        float acc = 0.0f;
        for (int n = 0; n < N_; ++n) acc += s_wr[n] * s_mem[n * MLD + tid];
        r_out[b * M_ + tid] = acc;
    }

    // memory update, coalesced store
    float* mo = mem_out + b * (size_t)(N_ * M_);
    for (int i = 0; i < 64; ++i) {
        int idx = i * 128 + tid;
        int row = idx >> 6, cm = idx & 63;
        float v  = s_mem[row * MLD + cm];
        float wv = s_ww[row];
        mo[idx] = v * (1.0f - wv * s_e[cm]) + wv * s_a[cm];
    }
}

// ---------------------------------------------------------------------------
// Kernel 4: o = [h, r] @ fc_W.T + fc_b   (K = 320, 8 output tiles / 8 waves)
// ---------------------------------------------------------------------------
__global__ __launch_bounds__(256) void k_output(
    const float* __restrict__ h_in, const float* __restrict__ r_in,
    const _Float16* __restrict__ fcW_h, const float* __restrict__ fc_b,
    float* __restrict__ o_out)
{
    constexpr int A_LD = 328;   // 320 + pad (mult of 8)
    __shared__ _Float16 sA[16 * A_LD];

    int tid = threadIdx.x;
    int b0  = blockIdx.x * 16;

    for (int idx = tid; idx < 16 * (C_ + M_); idx += 256) {
        int r = idx / (C_ + M_), c = idx % (C_ + M_);
        float v = (c < C_) ? h_in[(size_t)(b0 + r) * C_ + c]
                           : r_in[(size_t)(b0 + r) * M_ + (c - C_)];
        sA[r * A_LD + c] = (_Float16)v;
    }
    __syncthreads();

    int wv = tid >> 5, lane = tid & 31;
    int col = lane & 15, hi = lane >> 4;
    int jt = wv * 16;

    v8f acc = {};
    for (int kk = 0; kk < (C_ + M_); kk += 32) {
        v16h a = load_a_frag(sA, A_LD, kk, lane);
        acc = wmma_acc(a, load_b_frag(fcW_h, C_ + M_, jt, kk, lane), acc);
    }
    int j = jt + col;
    float bv = fc_b[j];
#pragma unroll
    for (int i = 0; i < 8; ++i)
        o_out[(size_t)(b0 + i + hi * 8) * DOUT_ + j] = acc[i] + bv;
}

// ---------------------------------------------------------------------------
// Launch
// ---------------------------------------------------------------------------
extern "C" void kernel_launch(void* const* d_in, const int* in_sizes, int n_in,
                              void* d_out, int out_size, void* d_ws, size_t ws_size,
                              hipStream_t stream)
{
    (void)in_sizes; (void)n_in; (void)out_size; (void)ws_size;

    const float* x            = (const float*)d_in[0];
    const float* prev_read    = (const float*)d_in[1];
    const float* h_prev       = (const float*)d_in[2];
    const float* w_read_prev  = (const float*)d_in[3];
    const float* w_write_prev = (const float*)d_in[4];
    const float* memory       = (const float*)d_in[5];
    const float* W_ih         = (const float*)d_in[6];
    const float* b_ih         = (const float*)d_in[7];
    const float* W_hh         = (const float*)d_in[8];
    const float* b_hh         = (const float*)d_in[9];
    const float* read_W       = (const float*)d_in[10];
    const float* read_b       = (const float*)d_in[11];
    const float* write_W      = (const float*)d_in[12];
    const float* write_b      = (const float*)d_in[13];
    const float* fc_W         = (const float*)d_in[14];
    const float* fc_b         = (const float*)d_in[15];

    // Output tuple laid out flat in return order
    float* out     = (float*)d_out;
    float* o_out   = out;
    float* h_out   = o_out  + (size_t)B_ * DOUT_;
    float* r_out   = h_out  + (size_t)B_ * C_;
    float* wr_out  = r_out  + (size_t)B_ * M_;
    float* ww_out  = wr_out + (size_t)B_ * N_;
    float* mem_out = ww_out + (size_t)B_ * N_;

    // Workspace layout (bytes):
    //   ro_s (B x 80 f32) | wo_s (B x 208 f32) | f16 weights | padded biases
    char* ws = (char*)d_ws;
    size_t off = 0;
    float* ro_s = (float*)(ws + off);  off += (size_t)B_ * RO_PAD * 4;      // 2.62 MB
    float* wo_s = (float*)(ws + off);  off += (size_t)B_ * WO_PAD * 4;      // 6.82 MB
    _Float16* Wih_h   = (_Float16*)(ws + off); off += (size_t)3 * C_ * CTRL_IN_ * 2;
    _Float16* Whh_h   = (_Float16*)(ws + off); off += (size_t)3 * C_ * C_ * 2;
    _Float16* readW_h = (_Float16*)(ws + off); off += (size_t)RO_PAD * C_ * 2;
    _Float16* writeW_h= (_Float16*)(ws + off); off += (size_t)WO_PAD * C_ * 2;
    _Float16* fcW_h   = (_Float16*)(ws + off); off += (size_t)DOUT_ * (C_ + M_) * 2;
    float* read_b_p   = (float*)(ws + off);    off += (size_t)RO_PAD * 4;
    float* write_b_p  = (float*)(ws + off);    off += (size_t)WO_PAD * 4;

    dim3 blk(256);
    dim3 grd(B_ / 16);
    k_cvt<<<dim3(256), blk, 0, stream>>>(W_ih, W_hh, read_W, read_b, write_W, write_b, fc_W,
                                         Wih_h, Whh_h, readW_h, writeW_h, fcW_h,
                                         read_b_p, write_b_p);
    k_controller<<<grd, blk, 0, stream>>>(x, prev_read, h_prev, Wih_h, b_ih, Whh_h, b_hh, h_out);
    k_heads<<<grd, blk, 0, stream>>>(h_out, readW_h, read_b_p, writeW_h, write_b_p, ro_s, wo_s);
    k_memory<<<dim3(B_), dim3(128), 0, stream>>>(memory, w_read_prev, w_write_prev,
                                                 ro_s, wo_s, r_out, wr_out, ww_out, mem_out);
    k_output<<<grd, blk, 0, stream>>>(h_out, r_out, fcW_h, fc_b, o_out);
}
